// WanAttentionBlock_66494683677332
// MI455X (gfx1250) — compile-verified, matrix operand order checked by reading, not run on hardware
//
#include <hip/hip_runtime.h>
#include <hip/hip_bf16.h>
#include <math.h>

// ---------------------------------------------------------------------------
// Types for CDNA5 WMMA (wave32, v_wmma_f32_16x16x32_bf16)
// ---------------------------------------------------------------------------
typedef __bf16 bf16_t;
typedef __attribute__((ext_vector_type(16))) __bf16 v16bf;
typedef __attribute__((ext_vector_type(8)))  __bf16 v8bf;
typedef __attribute__((ext_vector_type(8)))  float  v8f;
typedef __attribute__((ext_vector_type(4)))  float  v4f;

__device__ __forceinline__ v16bf cat8(v8bf lo, v8bf hi) {
    return __builtin_shufflevector(lo, hi, 0,1,2,3,4,5,6,7,8,9,10,11,12,13,14,15);
}
__device__ __forceinline__ v8f zero8() {
    v8f z = {0.f,0.f,0.f,0.f,0.f,0.f,0.f,0.f};
    return z;
}
__device__ __forceinline__ v8f wmma_bf16(v16bf a, v16bf b, v8f c) {
    return __builtin_amdgcn_wmma_f32_16x16x32_bf16(false, a, false, b, (short)0, c, false, false);
}
// reductions across the 16 lanes that share a C/D-layout row (wave32)
__device__ __forceinline__ float red_sum16(float v) {
    v += __shfl_xor(v, 1, 32); v += __shfl_xor(v, 2, 32);
    v += __shfl_xor(v, 4, 32); v += __shfl_xor(v, 8, 32);
    return v;
}
__device__ __forceinline__ float red_max16(float v) {
    v = fmaxf(v, __shfl_xor(v, 1, 32)); v = fmaxf(v, __shfl_xor(v, 2, 32));
    v = fmaxf(v, __shfl_xor(v, 4, 32)); v = fmaxf(v, __shfl_xor(v, 8, 32));
    return v;
}

// ---------------------------------------------------------------------------
// CDNA5 async global->LDS copy (ASYNCcnt-tracked; no VGPR round-trip).
// Generic LDS pointer's low 32 bits are the LDS byte address.
// Async ops complete IN ORDER, so s_wait_asynccnt N is a partial wait: with
// 4 ops per tile, waiting <=4 after issuing the next tile's 4 ops guarantees
// the previous tile has fully landed (double-buffer pipelining).
// ---------------------------------------------------------------------------
__device__ __forceinline__ void async_copy_b128(void* lds_ptr, const void* gptr) {
    uint32_t loff = (uint32_t)(uintptr_t)lds_ptr;
    uint64_t gaddr = (uint64_t)(uintptr_t)gptr;
    asm volatile("global_load_async_to_lds_b128 %0, %1, off"
                 :: "v"(loff), "v"(gaddr) : "memory");
}
__device__ __forceinline__ void wait_async0() {
    asm volatile("s_wait_asynccnt 0" ::: "memory");
}
__device__ __forceinline__ void wait_async4() {
    asm volatile("s_wait_asynccnt 4" ::: "memory");
}

// ---------------------------------------------------------------------------
// Problem constants (match reference)
// ---------------------------------------------------------------------------
#define SEQ   2560
#define DIMC  2048
#define NHEAD 16
#define HDIM  128
#define FFNC  8192
#define CTXL  512
#define CTXD  4096
#define FF    4
#define HH    20
#define WW    32
#define CF    22   // rope split of HDIM/2 = 64: 22/21/21
#define CH    21
#define CW    21
#define EPSV  1e-6f

// ---------------------------------------------------------------------------
// tiny elementwise kernels
// ---------------------------------------------------------------------------
__global__ void add_f32(const float* a, const float* b, float* o, int n) {
    int i = blockIdx.x * blockDim.x + threadIdx.x;
    if (i < n) o[i] = a[i] + b[i];
}
__global__ void f32_to_bf16(const float* a, bf16_t* o, int n) {
    int i = blockIdx.x * blockDim.x + threadIdx.x;
    if (i < n) o[i] = (bf16_t)a[i];
}

// ---------------------------------------------------------------------------
// Weight prep: src f32 [K,N] -> dst bf16 [N,K] (transposed), 32x32 LDS tiles.
// Paid once per launch; afterwards every GEMM B tile is a contiguous b128
// async copy and all weight re-reads hit the 192MB L2 in bf16.
// ---------------------------------------------------------------------------
__global__ __launch_bounds__(256)
void transpose_f32_to_bf16(const float* __restrict__ src, bf16_t* __restrict__ dst,
                           int K, int N) {
    __shared__ float t[32][33];
    const int bn = blockIdx.x * 32;       // N tile
    const int bk = blockIdx.y * 32;       // K tile
    const int tx = threadIdx.x & 31, ty = threadIdx.x >> 5;  // 32 x 8
    #pragma unroll
    for (int u = 0; u < 4; u++)
        t[ty + u * 8][tx] = src[(size_t)(bk + ty + u * 8) * N + bn + tx];
    __syncthreads();
    #pragma unroll
    for (int u = 0; u < 4; u++)
        dst[(size_t)(bn + ty + u * 8) * K + bk + tx] = (bf16_t)t[tx][ty + u * 8];
}

// ---------------------------------------------------------------------------
// LayerNorm (2048 cols) with affine:  out = ln(x) * G + B,  G = g (+1 opt)
// one block (256 thr) per row, 8 elements per thread
// ---------------------------------------------------------------------------
__global__ __launch_bounds__(256)
void layernorm_affine_bf16(const float* __restrict__ x,
                           const float* __restrict__ g,
                           const float* __restrict__ b,
                           int add_one, bf16_t* __restrict__ out) {
    const int row = blockIdx.x;
    const int tid = threadIdx.x;
    const int lane = tid & 31, wave = tid >> 5;
    __shared__ float sh_s[8], sh_q[8];
    float xv[8];
    float s = 0.f, q = 0.f;
    const float* xr = x + (size_t)row * DIMC + tid * 8;
    #pragma unroll
    for (int u = 0; u < 8; u++) { xv[u] = xr[u]; s += xv[u]; q += xv[u] * xv[u]; }
    #pragma unroll
    for (int m = 1; m < 32; m <<= 1) { s += __shfl_xor(s, m, 32); q += __shfl_xor(q, m, 32); }
    if (lane == 0) { sh_s[wave] = s; sh_q[wave] = q; }
    __syncthreads();
    float ts = 0.f, tq = 0.f;
    #pragma unroll
    for (int w = 0; w < 8; w++) { ts += sh_s[w]; tq += sh_q[w]; }
    const float mean = ts * (1.f / DIMC);
    const float var  = tq * (1.f / DIMC) - mean * mean;
    const float rinv = rsqrtf(var + EPSV);
    bf16_t* orow = out + (size_t)row * DIMC + tid * 8;
    #pragma unroll
    for (int u = 0; u < 8; u++) {
        const int n = tid * 8 + u;
        const float G = add_one ? (1.f + g[n]) : g[n];
        orow[u] = (bf16_t)((xv[u] - mean) * rinv * G + b[n]);
    }
}

// ---------------------------------------------------------------------------
// RMSNorm (2048 cols) + optional 3D RoPE, bf16 out. One block per row s.
// ---------------------------------------------------------------------------
__global__ __launch_bounds__(256)
void rmsnorm_rope_bf16(const float* __restrict__ x,
                       const float* __restrict__ w,
                       const float* __restrict__ angf,
                       const float* __restrict__ angh,
                       const float* __restrict__ angw,
                       int use_rope, bf16_t* __restrict__ out) {
    const int row = blockIdx.x;
    const int tid = threadIdx.x;
    const int lane = tid & 31, wave = tid >> 5;
    __shared__ float sh_q[8];
    float xv[8];
    float q = 0.f;
    const float* xr = x + (size_t)row * DIMC + tid * 8;
    #pragma unroll
    for (int u = 0; u < 8; u++) { xv[u] = xr[u]; q += xv[u] * xv[u]; }
    #pragma unroll
    for (int m = 1; m < 32; m <<= 1) q += __shfl_xor(q, m, 32);
    if (lane == 0) sh_q[wave] = q;
    __syncthreads();
    float tq = 0.f;
    #pragma unroll
    for (int wv = 0; wv < 8; wv++) tq += sh_q[wv];
    const float rinv = rsqrtf(tq * (1.f / DIMC) + EPSV);

    const int fpos = row / (HH * WW);
    const int hpos = (row / WW) % HH;
    const int wpos = row % WW;

    float yv[8];
    #pragma unroll
    for (int u = 0; u < 8; u++) {
        const int n = tid * 8 + u;
        yv[u] = xv[u] * rinv * w[n];
    }
    bf16_t* orow = out + (size_t)row * DIMC + tid * 8;
    if (use_rope) {
        #pragma unroll
        for (int p = 0; p < 4; p++) {
            const int n = tid * 8 + 2 * p;       // even channel index
            const int j = (n & (HDIM - 1)) >> 1; // pair index within head, 0..63
            float ang;
            if (j < CF)            ang = angf[fpos * CF + j];
            else if (j < CF + CH)  ang = angh[hpos * CH + (j - CF)];
            else                   ang = angw[wpos * CW + (j - CF - CH)];
            const float c = __cosf(ang), s = __sinf(ang);
            const float xe = yv[2 * p], xo = yv[2 * p + 1];
            orow[2 * p]     = (bf16_t)(xe * c - xo * s);
            orow[2 * p + 1] = (bf16_t)(xe * s + xo * c);
        }
    } else {
        #pragma unroll
        for (int u = 0; u < 8; u++) orow[u] = (bf16_t)yv[u];
    }
}

// ---------------------------------------------------------------------------
// GEMM: out[M,N] = A_bf16[M,K] @ Bt_bf16[N,K]^T (+bias) with fused epilogues.
// Block 256 thr = 8 waves; tile 128x128, K-step 32. DOUBLE-BUFFERED:
// tile i+1 is async-staged (global_load_async_to_lds_b128) into the alternate
// LDS buffer before s_wait_asynccnt 4 releases tile i for compute, so the
// global latency is hidden under the 8 WMMAs of the current tile.
// epilogue: 0 f32+bias, 1 bf16+bias, 2 gelu->bf16, 3 resid + (acc+bias)*scale
// ---------------------------------------------------------------------------
#define LDT 40  // padded LDS row stride in bf16 (80B: b128-aligned, conflict-free)

__global__ __launch_bounds__(256)
void gemm_bf16_wmma(const bf16_t* __restrict__ A, const bf16_t* __restrict__ Bt,
                    const float* __restrict__ bias, const float* __restrict__ scale,
                    const float* __restrict__ resid,
                    float* __restrict__ outF, bf16_t* __restrict__ outB,
                    int M, int N, int K, int epilogue) {
    __shared__ bf16_t As[2][128 * LDT];
    __shared__ bf16_t Bs[2][128 * LDT];
    const int tid = threadIdx.x;
    const int lane = tid & 31, wave = tid >> 5;
    const int wrow = wave & 3;   // 4 x 32-row slices
    const int wcol = wave >> 2;  // 2 x 64-col slices
    const int bm = blockIdx.y * 128;
    const int bn = blockIdx.x * 128;
    const int half = lane >> 4, l15 = lane & 15;
    const int khalf = half * 8;

    v8f acc[2][4];
    #pragma unroll
    for (int i = 0; i < 2; i++)
        #pragma unroll
        for (int j = 0; j < 4; j++) acc[i][j] = zero8();

    const int sr = tid >> 1;          // staging row 0..127 (for both A and B)
    const int sc = (tid & 1) * 16;    // 16 bf16 (32B) per thread
    const bf16_t* arow = A  + (size_t)(bm + sr) * K + sc;
    const bf16_t* brow = Bt + (size_t)(bn + sr) * K + sc;

    // prologue: stage tile 0 into buffer 0 (4 async ops per thread)
    async_copy_b128(&As[0][sr * LDT + sc],     arow);
    async_copy_b128(&As[0][sr * LDT + sc + 8], arow + 8);
    async_copy_b128(&Bs[0][sr * LDT + sc],     brow);
    async_copy_b128(&Bs[0][sr * LDT + sc + 8], brow + 8);

    for (int k0 = 0; k0 < K; k0 += 32) {
        const int p = (k0 >> 5) & 1;
        // stage next tile into alternate buffer, then partial-wait current
        if (k0 + 32 < K) {
            const int kn = k0 + 32;
            async_copy_b128(&As[1 - p][sr * LDT + sc],     arow + kn);
            async_copy_b128(&As[1 - p][sr * LDT + sc + 8], arow + kn + 8);
            async_copy_b128(&Bs[1 - p][sr * LDT + sc],     brow + kn);
            async_copy_b128(&Bs[1 - p][sr * LDT + sc + 8], brow + kn + 8);
            wait_async4();   // in-order completion => tile k0 has landed
        } else {
            wait_async0();
        }
        __syncthreads();

        v16bf afrag[2], bfrag[4];
        #pragma unroll
        for (int i = 0; i < 2; i++) {
            const int r = wrow * 32 + i * 16 + l15;
            afrag[i] = cat8(*(const v8bf*)&As[p][r * LDT + khalf],
                            *(const v8bf*)&As[p][r * LDT + 16 + khalf]);
        }
        #pragma unroll
        for (int j = 0; j < 4; j++) {
            const int c = wcol * 64 + j * 16 + l15;
            bfrag[j] = cat8(*(const v8bf*)&Bs[p][c * LDT + khalf],
                            *(const v8bf*)&Bs[p][c * LDT + 16 + khalf]);
        }
        #pragma unroll
        for (int i = 0; i < 2; i++)
            #pragma unroll
            for (int j = 0; j < 4; j++)
                acc[i][j] = wmma_bf16(afrag[i], bfrag[j], acc[i][j]);
        __syncthreads();   // all waves done reading buf p before it is refilled
    }

    // ---- epilogue ----
    #pragma unroll
    for (int i = 0; i < 2; i++)
        #pragma unroll
        for (int j = 0; j < 4; j++)
            #pragma unroll
            for (int v = 0; v < 8; v++) {
                const int m = bm + wrow * 32 + i * 16 + v + half * 8;
                const int n = bn + wcol * 64 + j * 16 + l15;
                const size_t idx = (size_t)m * N + n;
                float val = acc[i][j][v] + (bias ? bias[n] : 0.f);
                if (epilogue == 0) {
                    outF[idx] = val;
                } else if (epilogue == 1) {
                    outB[idx] = (bf16_t)val;
                } else if (epilogue == 2) {
                    const float t = 0.7978845608028654f * (val + 0.044715f * val * val * val);
                    outB[idx] = (bf16_t)(0.5f * val * (1.f + tanhf(t)));
                } else { // 3: residual (+ optional per-column gate)
                    const float sc2 = scale ? scale[n] : 1.f;
                    outF[idx] = resid[idx] + val * sc2;
                }
            }
}

// ---------------------------------------------------------------------------
// Flash attention, all-WMMA. Block = 4 waves x 16 query rows (64 q rows).
// grid = (Sq/64, NHEAD). K tiles async-staged; V transposed in LDS so the
// P@V B-fragments are contiguous b128 loads. Online softmax per row.
// ---------------------------------------------------------------------------
#define KT_LD 136   // Kt row stride (bf16), 272B, 16B-aligned
#define VT_LD 40    // Vt/Ps row stride

__global__ __launch_bounds__(128)
void flash_attn_wmma(const bf16_t* __restrict__ Q, const bf16_t* __restrict__ Kb,
                     const bf16_t* __restrict__ Vb, bf16_t* __restrict__ O,
                     int Sk, float scl) {
    __shared__ bf16_t Kt[32 * KT_LD];        // [key][hd]
    __shared__ bf16_t Vt[HDIM * VT_LD];      // [hd][key] (transposed)
    __shared__ bf16_t Ps[4][16 * VT_LD];     // per-wave P scratch [qrow][key]
    const int tid = threadIdx.x;
    const int lane = tid & 31, wave = tid >> 5;
    const int head = blockIdx.y;
    const int q0 = blockIdx.x * 64 + wave * 16;
    const int half = lane >> 4, l15 = lane & 15;
    const int khalf = half * 8;
    const int D = NHEAD * HDIM;

    // Q fragments: 4 x (16x32) chunks over hdim, loaded once from global
    v16bf qf[4];
    #pragma unroll
    for (int c = 0; c < 4; c++) {
        const bf16_t* qp = Q + (size_t)(q0 + l15) * D + head * HDIM + c * 32;
        qf[c] = cat8(*(const v8bf*)(qp + khalf), *(const v8bf*)(qp + 16 + khalf));
    }

    float mrow[8], lrow[8];
    v8f oacc[8];
    #pragma unroll
    for (int v = 0; v < 8; v++) { mrow[v] = -1e30f; lrow[v] = 0.f; }
    #pragma unroll
    for (int t = 0; t < 8; t++) oacc[t] = zero8();

    const int kr = tid >> 2;         // staging: key row 0..31
    const int kc = (tid & 3) * 32;   // 32 hd per thread

    for (int k0 = 0; k0 < Sk; k0 += 32) {
        // ---- stage K via async b128; V with explicit transpose ----
        {
            const bf16_t* kp = Kb + (size_t)(k0 + kr) * D + head * HDIM + kc;
            #pragma unroll
            for (int u = 0; u < 4; u++)
                async_copy_b128(&Kt[kr * KT_LD + kc + u * 8], kp + u * 8);
            const bf16_t* vp = Vb + (size_t)(k0 + kr) * D + head * HDIM + kc;
            #pragma unroll
            for (int u = 0; u < 32; u++)
                Vt[(kc + u) * VT_LD + kr] = vp[u];
        }
        wait_async0();
        __syncthreads();

        // ---- S = Q @ K^T : two 16x16 key subtiles ----
        v8f s0 = zero8(), s1 = zero8();
        #pragma unroll
        for (int c = 0; c < 4; c++) {
            v16bf b0 = cat8(*(const v8bf*)&Kt[l15 * KT_LD + c * 32 + khalf],
                            *(const v8bf*)&Kt[l15 * KT_LD + c * 32 + 16 + khalf]);
            v16bf b1 = cat8(*(const v8bf*)&Kt[(16 + l15) * KT_LD + c * 32 + khalf],
                            *(const v8bf*)&Kt[(16 + l15) * KT_LD + c * 32 + 16 + khalf]);
            s0 = wmma_bf16(qf[c], b0, s0);
            s1 = wmma_bf16(qf[c], b1, s1);
        }

        // ---- online softmax per row (row = 16 lanes of one half) ----
        #pragma unroll
        for (int v = 0; v < 8; v++) {
            const float a0 = s0[v] * scl;
            const float a1 = s1[v] * scl;
            float mx = red_max16(fmaxf(a0, a1));
            const float mnew = fmaxf(mrow[v], mx);
            const float f = __expf(mrow[v] - mnew);
            const float p0 = __expf(a0 - mnew);
            const float p1 = __expf(a1 - mnew);
            lrow[v] = lrow[v] * f + red_sum16(p0 + p1);
            mrow[v] = mnew;
            Ps[wave][(v + half * 8) * VT_LD + l15]      = (bf16_t)p0;
            Ps[wave][(v + half * 8) * VT_LD + 16 + l15] = (bf16_t)p1;
            #pragma unroll
            for (int t = 0; t < 8; t++) oacc[t][v] *= f;
        }

        // ---- O += P @ V (P from per-wave LDS scratch; DS in-order per wave) ----
        v16bf pf = cat8(*(const v8bf*)&Ps[wave][l15 * VT_LD + khalf],
                        *(const v8bf*)&Ps[wave][l15 * VT_LD + 16 + khalf]);
        #pragma unroll
        for (int t = 0; t < 8; t++) {
            v16bf vf = cat8(*(const v8bf*)&Vt[(t * 16 + l15) * VT_LD + khalf],
                            *(const v8bf*)&Vt[(t * 16 + l15) * VT_LD + 16 + khalf]);
            oacc[t] = wmma_bf16(pf, vf, oacc[t]);
        }
        __syncthreads();
    }

    // ---- write O (bf16, feeds next GEMM) ----
    #pragma unroll
    for (int t = 0; t < 8; t++)
        #pragma unroll
        for (int v = 0; v < 8; v++) {
            const int m = q0 + v + half * 8;
            const int hd = t * 16 + l15;
            O[(size_t)m * D + head * HDIM + hd] = (bf16_t)(oacc[t][v] / lrow[v]);
        }
}

// ---------------------------------------------------------------------------
// host-side orchestration
// ---------------------------------------------------------------------------
extern "C" void kernel_launch(void* const* d_in, const int* in_sizes, int n_in,
                              void* d_out, int out_size, void* d_ws, size_t ws_size,
                              hipStream_t stream) {
    (void)in_sizes; (void)n_in; (void)out_size; (void)ws_size;
    const float* x      = (const float*)d_in[0];
    const float* e      = (const float*)d_in[1];
    const float* ctx    = (const float*)d_in[2];
    const float* q_w    = (const float*)d_in[6];
    const float* q_b    = (const float*)d_in[7];
    const float* k_w    = (const float*)d_in[8];
    const float* k_b    = (const float*)d_in[9];
    const float* v_w    = (const float*)d_in[10];
    const float* v_b    = (const float*)d_in[11];
    const float* o_w    = (const float*)d_in[12];
    const float* o_b    = (const float*)d_in[13];
    const float* nq_w   = (const float*)d_in[14];
    const float* nk_w   = (const float*)d_in[15];
    const float* cq_w   = (const float*)d_in[16];
    const float* cq_b   = (const float*)d_in[17];
    const float* ck_w   = (const float*)d_in[18];
    const float* ck_b   = (const float*)d_in[19];
    const float* cv_w   = (const float*)d_in[20];
    const float* cv_b   = (const float*)d_in[21];
    const float* co_w   = (const float*)d_in[22];
    const float* co_b   = (const float*)d_in[23];
    const float* cnq_w  = (const float*)d_in[24];
    const float* cnk_w  = (const float*)d_in[25];
    const float* n3_g   = (const float*)d_in[26];
    const float* n3_b   = (const float*)d_in[27];
    const float* w1     = (const float*)d_in[28];
    const float* b1     = (const float*)d_in[29];
    const float* w2     = (const float*)d_in[30];
    const float* b2     = (const float*)d_in[31];
    const float* mod    = (const float*)d_in[32];
    const float* angf   = (const float*)d_in[33];
    const float* angh   = (const float*)d_in[34];
    const float* angw   = (const float*)d_in[35];
    float* out = (float*)d_out;

    // workspace carve-up
    char* p = (char*)d_ws;
    auto carve = [&](size_t bytes) { void* r = (void*)p; p += (bytes + 255) & ~(size_t)255; return r; };
    float*  em    = (float*) carve(6 * DIMC * sizeof(float));
    bf16_t* y_bf  = (bf16_t*)carve((size_t)SEQ * DIMC * 2);        // ln outputs (reused 3x)
    float*  pre1  = (float*) carve((size_t)SEQ * DIMC * 4);        // q_pre / ck_pre
    float*  pre2  = (float*) carve((size_t)SEQ * DIMC * 4);        // k_pre / cq_pre
    bf16_t* q_bf  = (bf16_t*)carve((size_t)SEQ * DIMC * 2);
    bf16_t* k_bf  = (bf16_t*)carve((size_t)SEQ * DIMC * 2);
    bf16_t* v_bf  = (bf16_t*)carve((size_t)SEQ * DIMC * 2);
    bf16_t* a_bf  = (bf16_t*)carve((size_t)SEQ * DIMC * 2);        // attn outputs
    float*  xbuf  = (float*) carve((size_t)SEQ * DIMC * 4);        // running residual
    bf16_t* c_bf  = (bf16_t*)carve((size_t)CTXL * CTXD * 2);       // context bf16
    bf16_t* h_bf  = (bf16_t*)carve((size_t)SEQ * FFNC * 2);        // ffn hidden
    // transposed bf16 weights [N,K]
    bf16_t* qw_t  = (bf16_t*)carve((size_t)DIMC * DIMC * 2);
    bf16_t* kw_t  = (bf16_t*)carve((size_t)DIMC * DIMC * 2);
    bf16_t* vw_t  = (bf16_t*)carve((size_t)DIMC * DIMC * 2);
    bf16_t* ow_t  = (bf16_t*)carve((size_t)DIMC * DIMC * 2);
    bf16_t* cqw_t = (bf16_t*)carve((size_t)DIMC * DIMC * 2);
    bf16_t* cow_t = (bf16_t*)carve((size_t)DIMC * DIMC * 2);
    bf16_t* ckw_t = (bf16_t*)carve((size_t)CTXD * DIMC * 2);
    bf16_t* cvw_t = (bf16_t*)carve((size_t)CTXD * DIMC * 2);
    bf16_t* w1_t  = (bf16_t*)carve((size_t)DIMC * FFNC * 2);
    bf16_t* w2_t  = (bf16_t*)carve((size_t)FFNC * DIMC * 2);

    const float attn_scale = 0.08838834764831845f; // 128^-0.5
    const dim3 gB(256);

    // ---- one-time (per launch) weight convert+transpose to bf16 [N,K] ----
    dim3 tD (DIMC / 32, DIMC / 32);
    dim3 tCK(DIMC / 32, CTXD / 32);
    dim3 tW1(FFNC / 32, DIMC / 32);
    dim3 tW2(DIMC / 32, FFNC / 32);
    transpose_f32_to_bf16<<<tD,  gB, 0, stream>>>(q_w,  qw_t,  DIMC, DIMC);
    transpose_f32_to_bf16<<<tD,  gB, 0, stream>>>(k_w,  kw_t,  DIMC, DIMC);
    transpose_f32_to_bf16<<<tD,  gB, 0, stream>>>(v_w,  vw_t,  DIMC, DIMC);
    transpose_f32_to_bf16<<<tD,  gB, 0, stream>>>(o_w,  ow_t,  DIMC, DIMC);
    transpose_f32_to_bf16<<<tD,  gB, 0, stream>>>(cq_w, cqw_t, DIMC, DIMC);
    transpose_f32_to_bf16<<<tD,  gB, 0, stream>>>(co_w, cow_t, DIMC, DIMC);
    transpose_f32_to_bf16<<<tCK, gB, 0, stream>>>(ck_w, ckw_t, CTXD, DIMC);
    transpose_f32_to_bf16<<<tCK, gB, 0, stream>>>(cv_w, cvw_t, CTXD, DIMC);
    transpose_f32_to_bf16<<<tW1, gB, 0, stream>>>(w1,   w1_t,  DIMC, FFNC);
    transpose_f32_to_bf16<<<tW2, gB, 0, stream>>>(w2,   w2_t,  FFNC, DIMC);

    // em = mod + e
    add_f32<<<dim3((6 * DIMC + 255) / 256), gB, 0, stream>>>(mod, e, em, 6 * DIMC);
    // y = ln(x)*(1+e1)+e0
    layernorm_affine_bf16<<<dim3(SEQ), gB, 0, stream>>>(x, em + 1 * DIMC, em + 0 * DIMC, 1, y_bf);

    // self-attn projections
    dim3 g2048(DIMC / 128, SEQ / 128);
    gemm_bf16_wmma<<<g2048, gB, 0, stream>>>(y_bf, qw_t, q_b, nullptr, nullptr, pre1, nullptr, SEQ, DIMC, DIMC, 0);
    gemm_bf16_wmma<<<g2048, gB, 0, stream>>>(y_bf, kw_t, k_b, nullptr, nullptr, pre2, nullptr, SEQ, DIMC, DIMC, 0);
    gemm_bf16_wmma<<<g2048, gB, 0, stream>>>(y_bf, vw_t, v_b, nullptr, nullptr, nullptr, v_bf, SEQ, DIMC, DIMC, 1);
    rmsnorm_rope_bf16<<<dim3(SEQ), gB, 0, stream>>>(pre1, nq_w, angf, angh, angw, 1, q_bf);
    rmsnorm_rope_bf16<<<dim3(SEQ), gB, 0, stream>>>(pre2, nk_w, angf, angh, angw, 1, k_bf);

    // self attention
    flash_attn_wmma<<<dim3(SEQ / 64, NHEAD), dim3(128), 0, stream>>>(q_bf, k_bf, v_bf, a_bf, SEQ, attn_scale);
    // x = x + (a@o_w + o_b) * e2
    gemm_bf16_wmma<<<g2048, gB, 0, stream>>>(a_bf, ow_t, o_b, em + 2 * DIMC, x, xbuf, nullptr, SEQ, DIMC, DIMC, 3);

    // cross attention
    layernorm_affine_bf16<<<dim3(SEQ), gB, 0, stream>>>(xbuf, n3_g, n3_b, 0, y_bf);
    f32_to_bf16<<<dim3((CTXL * CTXD + 255) / 256), gB, 0, stream>>>(ctx, c_bf, CTXL * CTXD);
    gemm_bf16_wmma<<<g2048, gB, 0, stream>>>(y_bf, cqw_t, cq_b, nullptr, nullptr, pre2, nullptr, SEQ, DIMC, DIMC, 0);
    dim3 gctx(DIMC / 128, CTXL / 128);
    gemm_bf16_wmma<<<gctx, gB, 0, stream>>>(c_bf, ckw_t, ck_b, nullptr, nullptr, pre1, nullptr, CTXL, DIMC, CTXD, 0);
    gemm_bf16_wmma<<<gctx, gB, 0, stream>>>(c_bf, cvw_t, cv_b, nullptr, nullptr, nullptr, v_bf, CTXL, DIMC, CTXD, 1);
    rmsnorm_rope_bf16<<<dim3(SEQ),  gB, 0, stream>>>(pre2, cnq_w, angf, angh, angw, 0, q_bf);
    rmsnorm_rope_bf16<<<dim3(CTXL), gB, 0, stream>>>(pre1, cnk_w, angf, angh, angw, 0, k_bf);
    flash_attn_wmma<<<dim3(SEQ / 64, NHEAD), dim3(128), 0, stream>>>(q_bf, k_bf, v_bf, a_bf, CTXL, attn_scale);
    // x = x + ca@co_w + co_b   (in-place residual)
    gemm_bf16_wmma<<<g2048, gB, 0, stream>>>(a_bf, cow_t, co_b, nullptr, xbuf, xbuf, nullptr, SEQ, DIMC, DIMC, 3);

    // FFN
    layernorm_affine_bf16<<<dim3(SEQ), gB, 0, stream>>>(xbuf, em + 4 * DIMC, em + 3 * DIMC, 1, y_bf);
    dim3 gffn1(FFNC / 128, SEQ / 128);
    gemm_bf16_wmma<<<gffn1, gB, 0, stream>>>(y_bf, w1_t, b1, nullptr, nullptr, nullptr, h_bf, SEQ, FFNC, DIMC, 2);
    // out = x + (gelu_h @ w2 + b2) * e5
    gemm_bf16_wmma<<<g2048, gB, 0, stream>>>(h_bf, w2_t, b2, em + 5 * DIMC, xbuf, out, nullptr, SEQ, DIMC, FFNC, 3);
}
